// MultiHeadAttention_7224134991903
// MI455X (gfx1250) — matrix operand order repaired and checked by reading, not compiled, not run on hardware
//
#include <hip/hip_runtime.h>

typedef __attribute__((ext_vector_type(4)))  _Float16 v4h;
typedef __attribute__((ext_vector_type(8)))  _Float16 v8h;
typedef __attribute__((ext_vector_type(16))) _Float16 v16h;
typedef __attribute__((ext_vector_type(8)))  float    v8f;

#define HEADS    8
#define HD       64
#define DIM      512
#define SEQ      2048
#define KTILE    64
#define NTILES   (SEQ / KTILE)
#define LDSPAD   72   // padded row stride in halves (144B, 16B-aligned)
// fold softmax scale AND log2(e) into Q so scores land in the exp2 domain
#define QSCALE   (0.125f * 1.44269504088896340736f)

// ---- fused DPP16 butterfly steps: one VALU op per step, no DS traffic ----
// step: v = op(dpp_perm(v), v)   (DPP applies to src0 only)
#define DPP_MAX_STEP(var, ctrl)                                              \
    asm("v_max_num_f32_dpp %0, %1, %1 " ctrl                                 \
        " row_mask:0xf bank_mask:0xf bound_ctrl:1"                           \
        : "=v"(var) : "v"(var))
#define DPP_ADD_STEP(var, ctrl)                                              \
    asm("v_add_f32_dpp %0, %1, %1 " ctrl                                     \
        " row_mask:0xf bank_mask:0xf bound_ctrl:1"                           \
        : "=v"(var) : "v"(var))

// all-reduce within each 16-lane group:
// xor1 (quad_perm), xor2 (quad_perm), ^7 (row_half_mirror), ^15 (row_mirror)
__device__ __forceinline__ float red_max16(float v) {
    DPP_MAX_STEP(v, "quad_perm:[1,0,3,2]");
    DPP_MAX_STEP(v, "quad_perm:[2,3,0,1]");
    DPP_MAX_STEP(v, "row_half_mirror");
    DPP_MAX_STEP(v, "row_mirror");
    return v;
}
__device__ __forceinline__ float red_sum16(float v) {
    DPP_ADD_STEP(v, "quad_perm:[1,0,3,2]");
    DPP_ADD_STEP(v, "quad_perm:[2,3,0,1]");
    DPP_ADD_STEP(v, "row_half_mirror");
    DPP_ADD_STEP(v, "row_mirror");
    return v;
}

__global__ __launch_bounds__(256, 1)
void fa_fwd_gfx1250(const float* __restrict__ Q, const float* __restrict__ K,
                    const float* __restrict__ V, float* __restrict__ O)
{
    __shared__ _Float16 Ksh[2][KTILE * LDSPAD];   // double-buffered [key][dim] f16
    __shared__ _Float16 Vsh[2][HD * LDSPAD];      // double-buffered [dim][key] f16
    __shared__ _Float16 Psh[8][16 * LDSPAD];      // per-wave P staging (C->A relayout)

    const int tid  = threadIdx.x;
    const int wave = tid >> 5;
    const int lane = tid & 31;
    const int ln   = lane & 15;   // n / row index within 16-group
    const int hl   = lane >> 4;   // which 16-lane half

    const int b  = blockIdx.z;
    const int h  = blockIdx.y;
    const int q0 = blockIdx.x * 128 + wave * 16;

    const float* Qb = Q + ((size_t)b * SEQ) * DIM + h * HD;
    const float* Kb = K + ((size_t)b * SEQ) * DIM + h * HD;
    const float* Vb = V + ((size_t)b * SEQ) * DIM + h * HD;
    float*       Ob = O + ((size_t)b * SEQ) * DIM + h * HD;

    // ---- Q (16 rows x 64 dims) -> two 16x32 A-layout f16 fragments, pre-scaled ----
    v16h qa[2];
    {
        const float* qrow = Qb + (size_t)(q0 + ln) * DIM;
        #pragma unroll
        for (int f = 0; f < 2; ++f) {
            const int k0 = 32 * f + 8 * hl;                     // halves 0..7
            const float4 a0 = *(const float4*)(qrow + k0);
            const float4 a1 = *(const float4*)(qrow + k0 + 4);
            const float4 a2 = *(const float4*)(qrow + k0 + 16); // halves 8..15
            const float4 a3 = *(const float4*)(qrow + k0 + 20);
            qa[f][0]=(_Float16)(a0.x*QSCALE); qa[f][1]=(_Float16)(a0.y*QSCALE);
            qa[f][2]=(_Float16)(a0.z*QSCALE); qa[f][3]=(_Float16)(a0.w*QSCALE);
            qa[f][4]=(_Float16)(a1.x*QSCALE); qa[f][5]=(_Float16)(a1.y*QSCALE);
            qa[f][6]=(_Float16)(a1.z*QSCALE); qa[f][7]=(_Float16)(a1.w*QSCALE);
            qa[f][8]=(_Float16)(a2.x*QSCALE); qa[f][9]=(_Float16)(a2.y*QSCALE);
            qa[f][10]=(_Float16)(a2.z*QSCALE); qa[f][11]=(_Float16)(a2.w*QSCALE);
            qa[f][12]=(_Float16)(a3.x*QSCALE); qa[f][13]=(_Float16)(a3.y*QSCALE);
            qa[f][14]=(_Float16)(a3.z*QSCALE); qa[f][15]=(_Float16)(a3.w*QSCALE);
        }
    }

    // ---- cooperative staging helpers (software pipeline) ----
    const int srow = tid >> 2;            // key 0..63 within tile
    const int scg  = (tid & 3) * 16;      // dim group
    const float* kbase = Kb + (size_t)srow * DIM + scg;
    const float* vbase = Vb + (size_t)srow * DIM + scg;

    float4 kreg[4], vreg[4];
    auto load_tile = [&](int jt) {
        const float* kr = kbase + (size_t)jt * DIM;
        const float* vr = vbase + (size_t)jt * DIM;
        #pragma unroll
        for (int g = 0; g < 4; ++g) {
            kreg[g] = *(const float4*)(kr + 4 * g);
            vreg[g] = *(const float4*)(vr + 4 * g);
        }
    };
    auto store_tile = [&](int bi) {
        #pragma unroll
        for (int g = 0; g < 4; ++g) {
            v4h hk = { (_Float16)kreg[g].x, (_Float16)kreg[g].y,
                       (_Float16)kreg[g].z, (_Float16)kreg[g].w };
            *(v4h*)&Ksh[bi][srow * LDSPAD + scg + 4 * g] = hk;
            Vsh[bi][(scg + 4*g + 0) * LDSPAD + srow] = (_Float16)vreg[g].x;
            Vsh[bi][(scg + 4*g + 1) * LDSPAD + srow] = (_Float16)vreg[g].y;
            Vsh[bi][(scg + 4*g + 2) * LDSPAD + srow] = (_Float16)vreg[g].z;
            Vsh[bi][(scg + 4*g + 3) * LDSPAD + srow] = (_Float16)vreg[g].w;
        }
    };

    // ---- online-softmax state ----
    float mrow[8], lrow[8];
    v8f acc[4];
    #pragma unroll
    for (int r = 0; r < 8; ++r) { mrow[r] = -1e30f; lrow[r] = 0.0f; }
    #pragma unroll
    for (int c = 0; c < 4; ++c) acc[c] = (v8f){0,0,0,0,0,0,0,0};

    // prologue: stage tile 0
    load_tile(0);
    store_tile(0);
    __syncthreads();

    for (int t = 0; t < NTILES; ++t) {
        const int cur = t & 1;
        const int nxt = cur ^ 1;

        // issue next tile's global loads early (latency hidden under WMMAs)
        if (t + 1 < NTILES) load_tile((t + 1) * KTILE);
        if (t + 2 < NTILES) {   // warm GL2 two tiles ahead (global_prefetch_b8)
            __builtin_prefetch(kbase + (size_t)(t + 2) * KTILE * DIM, 0, 1);
            __builtin_prefetch(vbase + (size_t)(t + 2) * KTILE * DIM, 0, 1);
        }

        // ---- S = Q*K^T : 4 chunks of 16x16, K-dim 64 = 2 WMMAs each ----
        v8f s[4];
        #pragma unroll
        for (int c = 0; c < 4; ++c) {
            v8f cz = (v8f){0,0,0,0,0,0,0,0};
            #pragma unroll
            for (int f = 0; f < 2; ++f) {
                const _Float16* bp = &Ksh[cur][(16*c + ln) * LDSPAD + 32*f + 8*hl];
                v8h lo = *(const v8h*)bp;
                v8h hi = *(const v8h*)(bp + 16);
                v16h bb;
                #pragma unroll
                for (int i = 0; i < 8; ++i) { bb[i] = lo[i]; bb[i+8] = hi[i]; }
                cz = __builtin_amdgcn_wmma_f32_16x16x32_f16(false, qa[f], false, bb,
                                                            (short)0, cz, false, false);
            }
            s[c] = cz;
        }

        // ---- online softmax in exp2 domain (scale pre-folded into Q) ----
        float alpha[8];
        #pragma unroll
        for (int r = 0; r < 8; ++r) {
            float v = fmaxf(fmaxf(s[0][r], s[1][r]), fmaxf(s[2][r], s[3][r]));
            v = red_max16(v);
            float mnew = fmaxf(mrow[r], v);
            alpha[r] = __builtin_exp2f(mrow[r] - mnew);
            mrow[r]  = mnew;
        }
        #pragma unroll
        for (int c = 0; c < 4; ++c)
            #pragma unroll
            for (int r = 0; r < 8; ++r)
                s[c][r] = __builtin_exp2f(s[c][r] - mrow[r]);
        #pragma unroll
        for (int r = 0; r < 8; ++r) {
            float v = (s[0][r] + s[1][r]) + (s[2][r] + s[3][r]);
            v = red_sum16(v);
            lrow[r] = lrow[r] * alpha[r] + v;
        }
        #pragma unroll
        for (int c = 0; c < 4; ++c)
            #pragma unroll
            for (int r = 0; r < 8; ++r) acc[c][r] *= alpha[r];

        // ---- P: C-layout f32 -> per-wave LDS -> A-layout f16 fragments ----
        _Float16* pp = Psh[wave];
        #pragma unroll
        for (int c = 0; c < 4; ++c)
            #pragma unroll
            for (int r = 0; r < 8; ++r)
                pp[(r + 8*hl) * LDSPAD + 16*c + ln] = (_Float16)s[c][r];
        // same-wave DS ops are in-order: wave-private buffer needs no barrier
        v16h pa[2];
        #pragma unroll
        for (int f = 0; f < 2; ++f) {
            const _Float16* ap = &pp[ln * LDSPAD + 32*f + 8*hl];
            v8h lo = *(const v8h*)ap;
            v8h hi = *(const v8h*)(ap + 16);
            #pragma unroll
            for (int i = 0; i < 8; ++i) { pa[f][i] = lo[i]; pa[f][i+8] = hi[i]; }
        }

        // ---- O += P*V : 4 dim-chunks x 2 key-fragments ----
        #pragma unroll
        for (int c = 0; c < 4; ++c) {
            #pragma unroll
            for (int f = 0; f < 2; ++f) {
                const _Float16* bp = &Vsh[cur][(16*c + ln) * LDSPAD + 32*f + 8*hl];
                v8h lo = *(const v8h*)bp;
                v8h hi = *(const v8h*)(bp + 16);
                v16h bb;
                #pragma unroll
                for (int i = 0; i < 8; ++i) { bb[i] = lo[i]; bb[i+8] = hi[i]; }
                acc[c] = __builtin_amdgcn_wmma_f32_16x16x32_f16(false, pa[f], false, bb,
                                                                (short)0, acc[c], false, false);
            }
        }

        // ---- stage next tile into the other buffer; one barrier per iter ----
        if (t + 1 < NTILES) store_tile(nxt);
        __syncthreads();
    }

    // ---- normalize and store (coalesced b32 across lanes) ----
    #pragma unroll
    for (int r = 0; r < 8; ++r) {
        const float inv = 1.0f / lrow[r];
        const int m = r + 8 * hl;
        float* orow = Ob + (size_t)(q0 + m) * DIM;
        #pragma unroll
        for (int c = 0; c < 4; ++c)
            orow[16*c + ln] = acc[c][r] * inv;
    }
}

extern "C" void kernel_launch(void* const* d_in, const int* in_sizes, int n_in,
                              void* d_out, int out_size, void* d_ws, size_t ws_size,
                              hipStream_t stream) {
    (void)n_in; (void)out_size; (void)d_ws; (void)ws_size;
    const float* Q = (const float*)d_in[0];
    const float* K = (const float*)d_in[1];
    const float* V = (const float*)d_in[2];
    float* O = (float*)d_out;
    const int B = in_sizes[0] / (SEQ * DIM);
    dim3 grid(SEQ / 128, HEADS, B);
    fa_fwd_gfx1250<<<grid, dim3(256), 0, stream>>>(Q, K, V, O);
}
